// TreeLSTM_Verifier_44976897523967
// MI455X (gfx1250) — matrix operand order repaired and checked by reading, not compiled, not run on hardware
//
#include <hip/hip_runtime.h>
#include <stdint.h>

#define NROWS 131072
#define DDIM  128
#define PITCH 136              // ushorts per LDS weight row (272 B = 68 dwords -> bank spread)
#define WSLICE (16 * PITCH)    // ushorts per weight slice (16 rows)
#define BUFELEMS (14 * WSLICE) // ushorts per buffer (14 weights)

typedef __attribute__((ext_vector_type(16))) __bf16          v16bf;
typedef __attribute__((ext_vector_type(16))) unsigned short  v16us;
typedef __attribute__((ext_vector_type(8)))  float           v8f;
typedef __attribute__((ext_vector_type(4)))  float           v4f;
typedef __attribute__((ext_vector_type(4)))  unsigned int    v4u;

static __device__ __forceinline__ unsigned short f32_to_bf16(float f) {
    unsigned int u = __builtin_bit_cast(unsigned int, f);
    u += 0x7FFFu + ((u >> 16) & 1u);   // round-to-nearest-even
    return (unsigned short)(u >> 16);
}

static __device__ __forceinline__ float fast_sigmoid(float x) {
    return 1.0f / (1.0f + __expf(-x));
}
static __device__ __forceinline__ float fast_tanh(float x) {
    float e = __expf(2.0f * x);        // overflow -> +1, underflow -> -1, no NaN
    return 1.0f - 2.0f / (e + 1.0f);
}

// ---------------- Kernel 0: fp32 weights -> bf16, transposed Wt[n][k] ----------------
struct WPtrs { const float* w[14]; };

__global__ __launch_bounds__(256)
void wconv_kernel(WPtrs wp, unsigned short* __restrict__ wt) {
    int idx = blockIdx.x * 256 + threadIdx.x;   // 14 * 128 * 128 total
    int w   = idx >> 14;
    int rem = idx & 16383;                      // rem = k*128 + n (row-major source)
    int k   = rem >> 7;
    int n   = rem & 127;
    wt[(w << 14) + (n << 7) + k] = f32_to_bf16(wp.w[w][rem]);
}

// ---------------- Kernel 1: fused 14-GEMM TreeLSTM cell ----------------
__global__ __launch_bounds__(256)
void treelstm_kernel(const float* __restrict__ hl, const float* __restrict__ cl,
                     const float* __restrict__ hr, const float* __restrict__ cr,
                     const float* __restrict__ p,
                     const unsigned short* __restrict__ wt,
                     const float* __restrict__ b_data, const float* __restrict__ b_fl,
                     const float* __restrict__ b_fr,  const float* __restrict__ b_out,
                     const float* __restrict__ b_in,
                     float* __restrict__ out_h, float* __restrict__ out_c) {
    // Double-buffered LDS stage for the per-nt weight slice (14 x 16 x 128 bf16, padded).
    __shared__ __align__(16) unsigned short wlds[2][BUFELEMS];

    const int tid  = threadIdx.x;
    const int wave = tid >> 5;          // 8 waves / block
    const int lane = tid & 31;
    const int m    = lane & 15;
    const int hi   = lane >> 4;
    const int row0 = blockIdx.x * 128 + wave * 16;

    // ---- async DMA of weight slice nt into wlds[buf]: wave v copies weights v and v+8 ----
    auto issue_slice = [&](int buf, int nt) {
#pragma unroll
        for (int t = 0; t < 2; ++t) {
            const int w = wave + t * 8;
            if (w < 14) {
                const unsigned long long gbase =
                    (unsigned long long)(const void*)(wt + ((size_t)w << 14) + (size_t)nt * 2048);
#pragma unroll
                for (int i = 0; i < 8; ++i) {   // 8 instrs x 512B = 4KB slice per weight
                    unsigned long long ga = gbase + (unsigned long long)(i * 512 + lane * 16);
                    unsigned int la = (unsigned int)(uintptr_t)
                        &wlds[buf][w * WSLICE + (2 * i + hi) * PITCH + m * 8];
                    asm volatile("global_load_async_to_lds_b128 %0, %1, off"
                                 :: "v"(la), "v"(ga) : "memory");
                }
            }
        }
    };

    // ---- A fragments: 16x32 bf16 tile of X rows [row0..row0+15], K-step ks ----
    auto load_afrag = [&](const float* __restrict__ X, int ks) -> v16bf {
        const float* base = X + (size_t)(row0 + m) * DDIM + ks * 32 + hi * 8;
        v4f a0 = *(const v4f*)(base + 0);
        v4f a1 = *(const v4f*)(base + 4);
        v4f a2 = *(const v4f*)(base + 16);
        v4f a3 = *(const v4f*)(base + 20);
        v16us t;
#pragma unroll
        for (int j = 0; j < 4; ++j) {
            t[j + 0]  = f32_to_bf16(a0[j]);
            t[j + 4]  = f32_to_bf16(a1[j]);
            t[j + 8]  = f32_to_bf16(a2[j]);
            t[j + 12] = f32_to_bf16(a3[j]);
        }
        return __builtin_bit_cast(v16bf, t);
    };

    // ---- B fragments from LDS: lane = column n, hi selects K half ----
    auto load_bfrag = [&](int buf, int w, int ks) -> v16bf {
        const unsigned short* src = &wlds[buf][(w * 16 + m) * PITCH + ks * 32 + hi * 8];
        union { v4u u[2]; v16bf v; } c;
        c.u[0] = *(const v4u*)(src);
        c.u[1] = *(const v4u*)(src + 16);
        return c.v;
    };

    auto mma = [](v8f c, v16bf a, v16bf b) -> v8f {
        return __builtin_amdgcn_wmma_f32_16x16x32_bf16(false, a, false, b,
                                                       (short)0, c, false, false);
    };

    // Keep all A fragments resident: 3 inputs x 4 K-steps x 8 VGPRs = 96 VGPRs
    v16bf ap[4], al[4], ar[4];
#pragma unroll
    for (int ks = 0; ks < 4; ++ks) {
        ap[ks] = load_afrag(p,  ks);
        al[ks] = load_afrag(hl, ks);
        ar[ks] = load_afrag(hr, ks);
    }

    issue_slice(0, 0);   // prime the pipeline

#pragma unroll 1
    for (int nt = 0; nt < 8; ++nt) {
        const int cur = nt & 1;

        // slice nt landed; every wave is also done reading buffer (nt+1)&1 from iter nt-1
        asm volatile("s_wait_asynccnt 0x0" ::: "memory");
        __syncthreads();

        if (nt < 7) issue_slice(cur ^ 1, nt + 1);   // overlap DMA of nt+1 with compute of nt

        // prefetch the cl/cr tile needed by the epilogue
        __builtin_prefetch(cl + (size_t)(row0 + m) * DDIM + nt * 16, 0, 3);
        __builtin_prefetch(cr + (size_t)(row0 + m) * DDIM + nt * 16, 0, 3);

        v8f acc_i  = {0.f,0.f,0.f,0.f,0.f,0.f,0.f,0.f};
        v8f acc_fp = acc_i, acc_fl = acc_i, acc_fr = acc_i, acc_o = acc_i, acc_u = acc_i;

#pragma unroll
        for (int ks = 0; ks < 4; ++ks) {
            acc_i  = mma(acc_i,  ap[ks], load_bfrag(cur,  0, ks)); // W_data
            acc_i  = mma(acc_i,  al[ks], load_bfrag(cur,  1, ks)); // W_data_l
            acc_i  = mma(acc_i,  ar[ks], load_bfrag(cur,  2, ks)); // W_data_r
            acc_fp = mma(acc_fp, ap[ks], load_bfrag(cur,  3, ks)); // W_forget
            acc_fl = mma(acc_fl, al[ks], load_bfrag(cur,  4, ks)); // W_fll
            acc_fl = mma(acc_fl, ar[ks], load_bfrag(cur,  5, ks)); // W_flr
            acc_fr = mma(acc_fr, al[ks], load_bfrag(cur,  6, ks)); // W_frl
            acc_fr = mma(acc_fr, ar[ks], load_bfrag(cur,  7, ks)); // W_frr
            acc_o  = mma(acc_o,  ap[ks], load_bfrag(cur,  8, ks)); // W_out
            acc_o  = mma(acc_o,  al[ks], load_bfrag(cur,  9, ks)); // W_out_l
            acc_o  = mma(acc_o,  ar[ks], load_bfrag(cur, 10, ks)); // W_out_r
            acc_u  = mma(acc_u,  ap[ks], load_bfrag(cur, 11, ks)); // W_in
            acc_u  = mma(acc_u,  al[ks], load_bfrag(cur, 12, ks)); // W_in_l
            acc_u  = mma(acc_u,  ar[ks], load_bfrag(cur, 13, ks)); // W_in_r
        }

        // ---- fused epilogue in D-matrix layout: col = nt*16 + m, row = row0 + hi*8 + r ----
        const int col = nt * 16 + m;
        const float bd  = b_data[col];
        const float bfl = b_fl[col];
        const float bfr = b_fr[col];
        const float bo  = b_out[col];
        const float bi  = b_in[col];

#pragma unroll
        for (int r = 0; r < 8; ++r) {
            const int row    = row0 + hi * 8 + r;
            const size_t off = (size_t)row * DDIM + col;
            const float iv  = fast_sigmoid(acc_i[r]  + bd);
            const float flv = fast_sigmoid(acc_fl[r] + acc_fp[r] + bfl);
            const float frv = fast_sigmoid(acc_fr[r] + acc_fp[r] + bfr);
            const float ov  = fast_sigmoid(acc_o[r]  + bo);
            const float uv  = fast_tanh(acc_u[r] + bi);
            const float cv  = iv * uv + flv * cl[off] + frv * cr[off];
            out_c[off] = cv;
            out_h[off] = ov * fast_tanh(cv);
        }
    }
}

extern "C" void kernel_launch(void* const* d_in, const int* in_sizes, int n_in,
                              void* d_out, int out_size, void* d_ws, size_t ws_size,
                              hipStream_t stream) {
    (void)in_sizes; (void)n_in; (void)out_size; (void)ws_size;

    const float* hl = (const float*)d_in[0];
    const float* cl = (const float*)d_in[1];
    const float* hr = (const float*)d_in[2];
    const float* cr = (const float*)d_in[3];
    const float* p  = (const float*)d_in[4];

    WPtrs wp;
    for (int j = 0; j < 14; ++j) wp.w[j] = (const float*)d_in[5 + j];

    const float* b_data = (const float*)d_in[19];
    const float* b_fl   = (const float*)d_in[20];
    const float* b_fr   = (const float*)d_in[21];
    const float* b_out  = (const float*)d_in[22];
    const float* b_in   = (const float*)d_in[23];

    unsigned short* wt = (unsigned short*)d_ws;   // 14 * 128 * 128 * 2B = 448 KB scratch
    float* out_h = (float*)d_out;
    float* out_c = out_h + (size_t)NROWS * DDIM;

    // Kernel 0: weight convert + transpose (14 * 16384 elements)
    wconv_kernel<<<dim3((14 * 16384) / 256), dim3(256), 0, stream>>>(wp, wt);

    // Kernel 1: fused TreeLSTM (1024 blocks * 8 waves, each wave = 16-row stripe)
    treelstm_kernel<<<dim3(NROWS / 128), dim3(256), 0, stream>>>(
        hl, cl, hr, cr, p, wt, b_data, b_fl, b_fr, b_out, b_in, out_h, out_c);
}
